// QuixerAttention_30339648979605
// MI455X (gfx1250) — matrix operand order repaired and checked by reading, not compiled, not run on hardware
//
#include <hip/hip_runtime.h>
#include <hip/hip_bf16.h>
#include <math.h>
#include <stdint.h>

typedef __attribute__((ext_vector_type(16))) __bf16 v16bf;
typedef __attribute__((ext_vector_type(8)))  __bf16 v8bf;
typedef __attribute__((ext_vector_type(8)))  float  v8f;

#define QB 2
#define QL 2048
#define QS 2048
#define QH 8
#define QE 64
#define NTOK (QB * QS * QH)   // 32768 tokens
#define NBS  (QB * QS)        // 4096 (b,s) pairs per (h,e) row of qevT
#define SM_SCALE 0.18033688f  // (1/sqrt(64)) * log2(e): softmax in base-2 logits

// ------------------------------------------------------------------
// DPP16 ROW_XMASK xor-swap within a 16-lane row (pure VALU, no LDS pipe).
// ------------------------------------------------------------------
template <int XMASK>
__device__ __forceinline__ float dpp_xor(float x) {
  return __builtin_bit_cast(float,
      __builtin_amdgcn_update_dpp(0, __builtin_bit_cast(int, x),
                                  0x160 | XMASK, 0xf, 0xf, true));
}
__device__ __forceinline__ float row_max16(float x) {
  x = fmaxf(x, dpp_xor<1>(x));
  x = fmaxf(x, dpp_xor<2>(x));
  x = fmaxf(x, dpp_xor<4>(x));
  x = fmaxf(x, dpp_xor<8>(x));
  return x;
}
__device__ __forceinline__ float row_sum16(float x) {
  x += dpp_xor<1>(x);
  x += dpp_xor<2>(x);
  x += dpp_xor<4>(x);
  x += dpp_xor<8>(x);
  return x;
}

// ------------------------------------------------------------------
// 4-qubit state helpers. wire w -> bit mask (8 >> w), amplitudes in
// registers (fully unrolled so indices fold to constants).
// ------------------------------------------------------------------
__device__ __forceinline__ void apply_ry(float* re, float* im, int mask, float th) {
  float c = __cosf(0.5f * th), s = __sinf(0.5f * th);
  #pragma unroll
  for (int i = 0; i < 16; i++) {
    if (i & mask) continue;
    int j = i | mask;
    float r0 = re[i], i0 = im[i], r1 = re[j], i1 = im[j];
    re[i] = c * r0 - s * r1;  im[i] = c * i0 - s * i1;
    re[j] = s * r0 + c * r1;  im[j] = s * i0 + c * i1;
  }
}

__device__ __forceinline__ void apply_crx(float* re, float* im, int cmask, int tmask, float th) {
  float c = __cosf(0.5f * th), s = __sinf(0.5f * th);
  #pragma unroll
  for (int i = 0; i < 16; i++) {
    if (!(i & cmask) || (i & tmask)) continue;  // control=1, target bit clear
    int j = i | tmask;
    float r0 = re[i], i0 = im[i], r1 = re[j], i1 = im[j];
    // a0' = c*a0 - i s*a1 ; a1' = -i s*a0 + c*a1
    re[i] = c * r0 + s * i1;  im[i] = c * i0 - s * r1;
    re[j] = c * r1 + s * i0;  im[j] = c * i1 - s * r0;
  }
}

// ------------------------------------------------------------------
// Kernel 1: quantum value encoding + K bf16 conversion.
// values [NTOK,64] f32 -> qevT [H][E][B*S] bf16 (ws, pre-transposed so
//                          attention V tiles are async-DMA friendly)
// keys   [NTOK,64] f32 -> Kb  [NTOK,64] bf16 (ws)
// ------------------------------------------------------------------
__global__ __launch_bounds__(128) void qev_kernel(
    const float* __restrict__ values, const float* __restrict__ keys,
    const float* __restrict__ W_ang,  const float* __restrict__ b_ang,
    const float* __restrict__ W_out,  const float* __restrict__ b_out,
    __bf16* __restrict__ qevT, __bf16* __restrict__ Kb)
{
  __shared__ float sv[128][QE + 1];   // +1 pad: conflict-free column reads
  __shared__ float swa[16][QE];
  __shared__ float swo[QE][12];
  __shared__ float sba[16];
  __shared__ float sbo[QE];

  const int tid  = threadIdx.x;
  const int tok0 = blockIdx.x * 128;

  // values for 128 tokens are contiguous in global memory
  for (int i = tid * 4; i < 128 * QE; i += 128 * 4) {
    float4 f = *(const float4*)&values[(size_t)tok0 * QE + i];
    int r = i >> 6, e = i & 63;
    sv[r][e] = f.x; sv[r][e + 1] = f.y; sv[r][e + 2] = f.z; sv[r][e + 3] = f.w;
    // fuse: convert same token range of K to bf16 for async staging later
    float4 g = *(const float4*)&keys[(size_t)tok0 * QE + i];
    __bf16* o = &Kb[(size_t)tok0 * QE + i];
    o[0] = (__bf16)g.x; o[1] = (__bf16)g.y; o[2] = (__bf16)g.z; o[3] = (__bf16)g.w;
  }
  for (int i = tid; i < 16 * QE; i += 128) swa[i >> 6][i & 63] = W_ang[i];
  for (int i = tid; i < QE * 12; i += 128) swo[i / 12][i % 12] = W_out[i];
  if (tid < 16) sba[tid] = b_ang[tid];
  if (tid < QE) sbo[tid] = b_out[tid];
  __syncthreads();

  // angles[p] = W_angles[p,:] . v + b_angles[p]
  float ang[16];
  #pragma unroll
  for (int p = 0; p < 16; p++) ang[p] = sba[p];
  for (int d = 0; d < QE; d++) {
    float vd = sv[tid][d];
    #pragma unroll
    for (int p = 0; p < 16; p++) ang[p] += swa[p][d] * vd;
  }

  // |0000> then Ansatz 14, one layer
  float re[16], im[16];
  #pragma unroll
  for (int i = 0; i < 16; i++) { re[i] = 0.f; im[i] = 0.f; }
  re[0] = 1.f;

  apply_ry(re, im, 8, ang[0]);  apply_ry(re, im, 4, ang[1]);
  apply_ry(re, im, 2, ang[2]);  apply_ry(re, im, 1, ang[3]);
  apply_crx(re, im, 1, 8, ang[4]);   // ctrl q3 -> tgt q0
  apply_crx(re, im, 2, 1, ang[5]);   // ctrl q2 -> tgt q3
  apply_crx(re, im, 4, 2, ang[6]);   // ctrl q1 -> tgt q2
  apply_crx(re, im, 8, 4, ang[7]);   // ctrl q0 -> tgt q1
  apply_ry(re, im, 8, ang[8]);  apply_ry(re, im, 4, ang[9]);
  apply_ry(re, im, 2, ang[10]); apply_ry(re, im, 1, ang[11]);
  apply_crx(re, im, 1, 2, ang[12]);  // ctrl q3 -> tgt q2
  apply_crx(re, im, 8, 1, ang[13]);  // ctrl q0 -> tgt q3
  apply_crx(re, im, 4, 8, ang[14]);  // ctrl q1 -> tgt q0
  apply_crx(re, im, 2, 4, ang[15]);  // ctrl q2 -> tgt q1

  // <X>, <Y>, <Z> per wire
  float meas[12];
  #pragma unroll
  for (int w = 0; w < 4; w++) {
    int mu = 8 >> w;
    float pr = 0.f, pi = 0.f, z = 0.f;
    #pragma unroll
    for (int i = 0; i < 16; i++) {
      if (i & mu) continue;
      int j = i | mu;
      pr += re[i] * re[j] + im[i] * im[j];
      pi += re[i] * im[j] - im[i] * re[j];
      z  += re[i] * re[i] + im[i] * im[i] - re[j] * re[j] - im[j] * im[j];
    }
    meas[w] = 2.f * pr; meas[4 + w] = 2.f * pi; meas[8 + w] = z;
  }

  // qevT[h][e][b*S+s] = meas . W_out[e,:] + b_out[e]
  const size_t tok = (size_t)tok0 + tid;   // (b*S + s)*H + h
  const int    h   = (int)(tok % QH);
  const size_t bs  = tok / QH;             // b*S + s
  #pragma unroll 4
  for (int e = 0; e < QE; e++) {
    float acc = sbo[e];
    #pragma unroll
    for (int j = 0; j < 12; j++) acc += meas[j] * swo[e][j];
    qevT[((size_t)h * QE + e) * NBS + bs] = (__bf16)acc;
  }
}

// ------------------------------------------------------------------
// Async DMA of one 64-row x 128B bf16 tile into LDS (row-major dest).
// 512 x 16B chunks spread over 128 threads; tracked by ASYNCcnt
// (4 instructions per wave per tile).
// ------------------------------------------------------------------
__device__ __forceinline__ void async_copy_tile(const __bf16* tile_base, size_t row_stride,
                                                uint32_t lds_base, int tid) {
  #pragma unroll
  for (int it = 0; it < 4; it++) {
    int idx = tid + it * 128;            // 0..511
    int r = idx >> 3, c = idx & 7;       // row, 16B chunk within row
    uint32_t lds = lds_base + (uint32_t)(r * 128 + c * 16);
    unsigned long long g =
        (unsigned long long)(uintptr_t)(tile_base + (size_t)r * row_stride + c * 8);
    asm volatile("global_load_async_to_lds_b128 %0, %1, off"
                 :: "v"(lds), "v"(g)
                 : "memory");
  }
}

// ------------------------------------------------------------------
// Kernel 2: causal flash attention with WMMA bf16, f32 accum.
// block = 128 threads (4 waves), 64 query rows per block, one (b,h)
// per blockIdx.y.  K and V tiles both arrive via async-to-LDS DMA,
// double-buffered: tile kb+1's DMA overlaps tile kb's WMMAs.
// ------------------------------------------------------------------
__global__ __launch_bounds__(128) void attn_kernel(
    const float* __restrict__ Q, const __bf16* __restrict__ Kb,
    const __bf16* __restrict__ VqT, float* __restrict__ Out)
{
  __shared__ __align__(32) __bf16 qs[64][QE];         // Q tile (row-major)
  __shared__ __align__(32) __bf16 ks[2][64][QE];      // K tiles [token][e] (async dest)
  __shared__ __align__(32) __bf16 vs[2][QE][64];      // V tiles [e][token]  (async dest)
  __shared__ __align__(32) __bf16 ps[4][16][64];      // per-wave P staging

  const int tid  = threadIdx.x;
  const int wave = tid >> 5, lane = tid & 31;
  const int l15  = lane & 15, grp = lane >> 4;        // lane half (C/A/B layouts)
  const int bh = blockIdx.y, b = bh >> 3, h = bh & 7;
  const int row0 = blockIdx.x * 64;
  const int bx   = (int)blockIdx.x;

  const __bf16* Ktile0 = &Kb[(((size_t)b * QS) * QH + h) * QE];         // +krow0*QH*QE per tile
  const __bf16* Vtile0 = &VqT[((size_t)h * QE) * NBS + (size_t)b * QS]; // +krow0 per tile

  // ---- stage Q tile, convert f32 -> bf16 ----
  for (int i = tid * 4; i < 64 * QE; i += 512) {
    int r = i >> 6, e = i & 63;
    float4 f = *(const float4*)&Q[((((size_t)b * QL) + row0 + r) * QH + h) * QE + e];
    qs[r][e] = (__bf16)f.x;  qs[r][e + 1] = (__bf16)f.y;
    qs[r][e + 2] = (__bf16)f.z; qs[r][e + 3] = (__bf16)f.w;
  }

  // ---- kick off async DMA of K/V tile 0 while Q settles ----
  async_copy_tile(Ktile0, QH * QE, (uint32_t)(uintptr_t)&ks[0][0][0], tid);
  async_copy_tile(Vtile0, NBS,     (uint32_t)(uintptr_t)&vs[0][0][0], tid);
  __syncthreads();

  // ---- persistent Q A-fragments (contiguous k = 16*half + s, 2 k-steps of 32) ----
  const int mrow = wave * 16 + l15;
  v16bf aq[2];
  #pragma unroll
  for (int kst = 0; kst < 2; kst++) {
    int lo = kst * 32 + grp * 16, hi = lo + 8;
    v8bf alo = *(const v8bf*)&qs[mrow][lo];
    v8bf ahi = *(const v8bf*)&qs[mrow][hi];
    #pragma unroll
    for (int j = 0; j < 8; j++) { aq[kst][j] = alo[j]; aq[kst][8 + j] = ahi[j]; }
  }

  v8f oacc[4];
  float mrun[8], lrun[8];
  #pragma unroll
  for (int nt = 0; nt < 4; nt++)
    #pragma unroll
    for (int j = 0; j < 8; j++) oacc[nt][j] = 0.f;
  #pragma unroll
  for (int r = 0; r < 8; r++) { mrun[r] = -3.0e38f; lrun[r] = 0.f; }

  // ---- key blocks up to and including the diagonal ----
  for (int kb = 0; kb <= bx; kb++) {
    const int krow0 = kb * 64;
    const int buf = kb & 1;

    // this wave's async writes for tile kb have landed
    asm volatile("s_wait_asynccnt 0x0" ::: "memory");
    __syncthreads();   // all waves' DMA visible; kb-1 buffers fully consumed

    // overlap: DMA the next K/V tiles into the other buffers while we compute
    if (kb < bx) {
      async_copy_tile(Ktile0 + (size_t)(krow0 + 64) * (QH * QE), QH * QE,
                      (uint32_t)(uintptr_t)&ks[buf ^ 1][0][0], tid);
      async_copy_tile(Vtile0 + (size_t)(krow0 + 64), NBS,
                      (uint32_t)(uintptr_t)&vs[buf ^ 1][0][0], tid);
    }

    // ---- preload all K B-fragments, then the 8 WMMAs (kst-outer: 4-apart accum) ----
    v16bf bk[4][2];
    #pragma unroll
    for (int nt = 0; nt < 4; nt++)
      #pragma unroll
      for (int kst = 0; kst < 2; kst++)
        bk[nt][kst] = *(const v16bf*)&ks[buf][nt * 16 + l15][kst * 32 + grp * 16];

    v8f sacc[4];
    #pragma unroll
    for (int nt = 0; nt < 4; nt++)
      #pragma unroll
      for (int j = 0; j < 8; j++) sacc[nt][j] = 0.f;
    #pragma unroll
    for (int kst = 0; kst < 2; kst++)
      #pragma unroll
      for (int nt = 0; nt < 4; nt++)
        sacc[nt] = __builtin_amdgcn_wmma_f32_16x16x32_bf16(false, aq[kst], false, bk[nt][kst],
                                                           (short)0, sacc[nt], false, false);

    // ---- causal mask (-1e9 pre-scale, as in reference), base-2 logits ----
    float tmax[8];
    #pragma unroll
    for (int r = 0; r < 8; r++) tmax[r] = -3.0e38f;
    #pragma unroll
    for (int nt = 0; nt < 4; nt++) {
      const int jg = krow0 + nt * 16 + l15;
      #pragma unroll
      for (int r = 0; r < 8; r++) {
        const int ig = row0 + wave * 16 + grp * 8 + r;  // C layout: VGPR r -> row r (+8 hi half)
        float s = sacc[nt][r];
        if (jg > ig) s = -1.0e9f;
        s *= SM_SCALE;                 // scale * log2(e): exp() -> exp2()
        sacc[nt][r] = s;
        tmax[r] = fmaxf(tmax[r], s);
      }
    }
    // row reductions across the 16-lane half via DPP ROW_XMASK (pure VALU)
    #pragma unroll
    for (int r = 0; r < 8; r++) tmax[r] = row_max16(tmax[r]);

    float fsc[8], rsum[8];
    #pragma unroll
    for (int r = 0; r < 8; r++) {
      float mnew = fmaxf(mrun[r], tmax[r]);
      fsc[r] = __builtin_amdgcn_exp2f(mrun[r] - mnew);
      mrun[r] = mnew;
      rsum[r] = 0.f;
    }
    #pragma unroll
    for (int nt = 0; nt < 4; nt++)
      #pragma unroll
      for (int r = 0; r < 8; r++) {
        float p = __builtin_amdgcn_exp2f(sacc[nt][r] - mrun[r]);
        sacc[nt][r] = p;
        rsum[r] += p;
      }
    #pragma unroll
    for (int r = 0; r < 8; r++) {
      lrun[r] = lrun[r] * fsc[r] + row_sum16(rsum[r]);
    }
    #pragma unroll
    for (int nt = 0; nt < 4; nt++)
      #pragma unroll
      for (int r = 0; r < 8; r++) oacc[nt][r] *= fsc[r];

    // ---- C-layout -> A-layout for P via per-wave LDS strip ----
    #pragma unroll
    for (int nt = 0; nt < 4; nt++)
      #pragma unroll
      for (int r = 0; r < 8; r++)
        ps[wave][grp * 8 + r][nt * 16 + l15] = (__bf16)sacc[nt][r];
    asm volatile("s_wait_dscnt 0" ::: "memory");   // same-wave LDS RAW fence

    v16bf ap[2];
    #pragma unroll
    for (int kst = 0; kst < 2; kst++) {
      int lo = kst * 32 + grp * 16, hi = lo + 8;
      v8bf alo = *(const v8bf*)&ps[wave][l15][lo];
      v8bf ahi = *(const v8bf*)&ps[wave][l15][hi];
      #pragma unroll
      for (int j = 0; j < 8; j++) { ap[kst][j] = alo[j]; ap[kst][8 + j] = ahi[j]; }
    }

    // ---- preload all V B-fragments, then O += P . V (8 WMMAs) ----
    v16bf bv[4][2];
    #pragma unroll
    for (int nt = 0; nt < 4; nt++)
      #pragma unroll
      for (int kst = 0; kst < 2; kst++)
        bv[nt][kst] = *(const v16bf*)&vs[buf][nt * 16 + l15][kst * 32 + grp * 16];
    #pragma unroll
    for (int kst = 0; kst < 2; kst++)
      #pragma unroll
      for (int nt = 0; nt < 4; nt++)
        oacc[nt] = __builtin_amdgcn_wmma_f32_16x16x32_bf16(false, ap[kst], false, bv[nt][kst],
                                                           (short)0, oacc[nt], false, false);
  }

  // ---- normalize and write out (f32, [B,L,H,E]) ----
  #pragma unroll
  for (int nt = 0; nt < 4; nt++)
    #pragma unroll
    for (int r = 0; r < 8; r++) {
      const int ig = row0 + wave * 16 + grp * 8 + r;
      Out[((((size_t)b * QL) + ig) * QH + h) * QE + nt * 16 + l15] =
          oacc[nt][r] / lrun[r];
    }
}

// ------------------------------------------------------------------
extern "C" void kernel_launch(void* const* d_in, const int* in_sizes, int n_in,
                              void* d_out, int out_size, void* d_ws, size_t ws_size,
                              hipStream_t stream) {
  const float* Q   = (const float*)d_in[0];
  const float* Kp  = (const float*)d_in[1];
  const float* V   = (const float*)d_in[2];
  const float* Wa  = (const float*)d_in[3];
  const float* ba  = (const float*)d_in[4];
  const float* Wo  = (const float*)d_in[5];
  const float* bo  = (const float*)d_in[6];
  float* Out = (float*)d_out;
  __bf16* qevT = (__bf16*)d_ws;                     // [H][E][B*S]  4 MB
  __bf16* Kb   = qevT + (size_t)NTOK * QE;          // [NTOK][E]    4 MB

  qev_kernel<<<dim3(NTOK / 128), dim3(128), 0, stream>>>(V, Kp, Wa, ba, Wo, bo, qevT, Kb);
  attn_kernel<<<dim3(QL / 64, QB * QH), dim3(128), 0, stream>>>(Q, Kb, qevT, Out);
}
